// TFOpenAIGPTMainLayer_60026462929543
// MI455X (gfx1250) — matrix operand order repaired and checked by reading, not compiled, not run on hardware
//
#include <hip/hip_runtime.h>

// ---- model constants (from reference) ----
#define BATCH   8
#define SEQ     512
#define TOKENS  4096        // BATCH*SEQ
#define DMODEL  768
#define NHEAD   12
#define BH      96          // BATCH*NHEAD
#define DHEAD   64
#define NLAYER  12
#define DFF     3072
#define QKVD    2304        // 3*DMODEL
#define LN_EPS  1e-5f

typedef __bf16 bf16_t;
typedef __attribute__((ext_vector_type(16))) __bf16 bf16x16;
typedef __attribute__((ext_vector_type(8)))  __bf16 bf16x8;
typedef __attribute__((ext_vector_type(8)))  float  f32x8;

#define SHUF16(lo, hi) __builtin_shufflevector(lo, hi, 0,1,2,3,4,5,6,7,8,9,10,11,12,13,14,15)

// ---------------- embedding: h = tok_emb[ids] + pos_emb ----------------
__global__ __launch_bounds__(256) void embed_kernel(const int* __restrict__ ids,
                                                    const float* __restrict__ tok,
                                                    const float* __restrict__ pos,
                                                    float* __restrict__ h) {
  size_t i = (size_t)blockIdx.x * 256 + threadIdx.x;
  if (i >= (size_t)TOKENS * DMODEL) return;
  int t = (int)(i / DMODEL), d = (int)(i % DMODEL);
  h[i] = tok[(size_t)ids[t] * DMODEL + d] + pos[(size_t)(t % SEQ) * DMODEL + d];
}

// ---------------- f32 -> bf16 (activations) ----------------
__global__ __launch_bounds__(256) void cvt_bf16_kernel(const float* __restrict__ src,
                                                       bf16_t* __restrict__ dst, size_t n) {
  size_t i = (size_t)blockIdx.x * 256 + threadIdx.x;
  if (i < n) dst[i] = (bf16_t)src[i];
}

// ------- f32 [L,K,N] -> bf16 transposed [L,N,K] (weights, once per call) -------
__global__ __launch_bounds__(256) void cvt_w_t_kernel(const float* __restrict__ src,
                                                      bf16_t* __restrict__ dst,
                                                      int K, int N, int nl) {
  size_t total = (size_t)nl * K * N;
  size_t i = (size_t)blockIdx.x * 256 + threadIdx.x;
  if (i >= total) return;
  size_t per = (size_t)K * N;
  size_t l = i / per, r = i % per;
  int k = (int)(r / N), n = (int)(r % N);
  dst[l * per + (size_t)n * K + k] = (bf16_t)src[i];
}

// ------- qkv [T,2304] f32 -> Qbf (x0.125) [BH,S,64], Kbf [BH,S,64], Vtbf [BH,64,S] -------
__global__ __launch_bounds__(256) void qkv_split_kernel(const float* __restrict__ qkv,
                                                        bf16_t* __restrict__ Qbf,
                                                        bf16_t* __restrict__ Kbf,
                                                        bf16_t* __restrict__ Vtbf) {
  size_t i = (size_t)blockIdx.x * 256 + threadIdx.x;
  if (i >= (size_t)TOKENS * DMODEL) return;
  int t = (int)(i / DMODEL), d = (int)(i % DMODEL);
  int b = t / SEQ, s = t % SEQ;
  int hh = d / DHEAD, dd = d % DHEAD;
  size_t z = (size_t)b * NHEAD + hh;
  const float* row = qkv + (size_t)t * QKVD;
  Qbf [(z * SEQ + s) * DHEAD + dd] = (bf16_t)(row[d] * 0.125f);   // 1/sqrt(64) folded in
  Kbf [(z * SEQ + s) * DHEAD + dd] = (bf16_t)row[DMODEL + d];
  Vtbf[(z * DHEAD + dd) * SEQ + s] = (bf16_t)row[2 * DMODEL + d];
}

__device__ __forceinline__ float gelu_tanh(float x) {
  return 0.5f * x * (1.0f + tanhf(0.7978845608028654f * (x + 0.044715f * x * x * x)));
}

// ================= weight GEMM: C[M,N] = A[M,K] @ Bt[N,K]^T + bias (opt GELU) =================
// 256 thr = 8 wave32, block tile 128x128, BK=32, double-buffered LDS.
// wave (wm 0..3, wn 0..1) owns 32 rows x 64 cols -> 2 A-frags x 4 B-frags -> 8 WMMA / K-step.
template <int ACT>
__global__ __launch_bounds__(256) void gemm_bf16_wmma(const bf16_t* __restrict__ A,
                                                      const bf16_t* __restrict__ Bt,
                                                      const float* __restrict__ bias,
                                                      float* __restrict__ C,
                                                      int M, int N, int K) {
  __shared__ bf16_t As[2][128][40];  // pitch 40: rows 16B-aligned, bank-conflict-free
  __shared__ bf16_t Bs[2][128][40];
  const int tid  = threadIdx.x;
  const int lane = tid & 31;
  const int wid  = tid >> 5;
  const int wm   = wid & 3;
  const int wn   = wid >> 2;
  const int m0 = blockIdx.y * 128, n0 = blockIdx.x * 128;
  const int lr = tid >> 1;          // coop-load row 0..127
  const int lc = (tid & 1) * 16;    // 2x uint4 (16 bf16) per thread per tile

  f32x8 acc[2][4];
#pragma unroll
  for (int i = 0; i < 2; ++i)
#pragma unroll
    for (int j = 0; j < 4; ++j) acc[i][j] = (f32x8){0.f,0.f,0.f,0.f,0.f,0.f,0.f,0.f};

  const int khalf = (lane >> 4) * 8;     // A frag: K {0..7,16..23} / {8..15,24..31}
  const int koff  = (lane >> 4) * 8;     // B frag: K {0..7,16..23} / {8..15,24..31}
  const int cl    = lane & 15;

  auto load_tile = [&](int kk, int buf_) {
    *(uint4*)&As[buf_][lr][lc]     = *(const uint4*)(A  + (size_t)(m0 + lr) * K + kk + lc);
    *(uint4*)&As[buf_][lr][lc + 8] = *(const uint4*)(A  + (size_t)(m0 + lr) * K + kk + lc + 8);
    *(uint4*)&Bs[buf_][lr][lc]     = *(const uint4*)(Bt + (size_t)(n0 + lr) * K + kk + lc);
    *(uint4*)&Bs[buf_][lr][lc + 8] = *(const uint4*)(Bt + (size_t)(n0 + lr) * K + kk + lc + 8);
  };

  int buf = 0;
  load_tile(0, 0);
  __syncthreads();

  for (int kk = 0; kk < K; kk += 32) {
    const int nx = buf ^ 1;
    if (kk + 32 < K) {
      load_tile(kk + 32, nx);                         // fill other buffer
      __builtin_prefetch(A  + (size_t)(m0 + lr) * K + kk + 64 + lc, 0, 1);
      __builtin_prefetch(Bt + (size_t)(n0 + lr) * K + kk + 64 + lc, 0, 1);
    }
    bf16x16 af[2];
#pragma unroll
    for (int i = 0; i < 2; ++i) {
      const int am = wm * 32 + i * 16 + cl;
      bf16x8 lo = *(const bf16x8*)&As[buf][am][khalf];
      bf16x8 hi = *(const bf16x8*)&As[buf][am][16 + khalf];
      af[i] = SHUF16(lo, hi);
    }
    bf16x16 bfr[4];
#pragma unroll
    for (int j = 0; j < 4; ++j) {
      const int bn = wn * 64 + j * 16 + cl;
      bf16x8 lo = *(const bf16x8*)&Bs[buf][bn][koff];
      bf16x8 hi = *(const bf16x8*)&Bs[buf][bn][koff + 16];
      bfr[j] = SHUF16(lo, hi);
    }
#pragma unroll
    for (int i = 0; i < 2; ++i)
#pragma unroll
      for (int j = 0; j < 4; ++j)
        acc[i][j] = __builtin_amdgcn_wmma_f32_16x16x32_bf16(false, af[i], false, bfr[j],
                                                            (short)0, acc[i][j], false, false);
    __syncthreads();
    buf = nx;
  }

  const int rbase = (lane >> 4) << 3;   // C/D layout: row = r (+8 upper half), col = lane&15
#pragma unroll
  for (int i = 0; i < 2; ++i) {
#pragma unroll
    for (int j = 0; j < 4; ++j) {
      const int mrow = m0 + wm * 32 + i * 16 + rbase;
      const int ncol = n0 + wn * 64 + j * 16 + cl;
      const float bb = bias[ncol];
#pragma unroll
      for (int r = 0; r < 8; ++r) {
        float v = acc[i][j][r] + bb;
        if (ACT == 1) v = gelu_tanh(v);
        C[(size_t)(mrow + r) * N + ncol] = v;
      }
    }
  }
}

// ============ batched attention GEMM (per z = b*H+h), tile 128x64, BK=32 ============
// MODE 0: scores[z] = Q[z] @ K[z]^T   (M=N=512, K=64)  -- causal upper-tile early-out
// MODE 1: a[..]     = P[z] @ Vt[z]^T  (M=512, N=64, K=512), scattered into [T,D]
template <int MODE>
__global__ __launch_bounds__(256) void attn_gemm_wmma(const bf16_t* __restrict__ Abase,
                                                      const bf16_t* __restrict__ Btbase,
                                                      float* __restrict__ Cbase,
                                                      int M, int N, int K) {
  const int z  = blockIdx.z;
  const int m0 = blockIdx.y * 128, n0 = blockIdx.x * 64;
  if (MODE == 0 && n0 > m0 + 127) return;   // tile fully above causal diagonal

  __shared__ bf16_t As[2][128][40];
  __shared__ bf16_t Bs[2][64][40];
  const int tid  = threadIdx.x;
  const int lane = tid & 31;
  const int wid  = tid >> 5;
  const int wm   = wid & 3;                 // 32-row strip
  const int wn   = wid >> 2;                // 32-col strip
  const int lra = tid >> 1,  lca = (tid & 1) * 16;  // A coop load: 128x32
  const int lrb = tid >> 2,  lcb = (tid & 3) * 8;   // B coop load: 64x32

  const bf16_t* A  = Abase  + (size_t)z * M * K;
  const bf16_t* Bt = Btbase + (size_t)z * N * K;
  float* C;
  int ldC;
  if (MODE == 0) { C = Cbase + (size_t)z * SEQ * SEQ; ldC = SEQ; }
  else           { C = Cbase + (size_t)(z / NHEAD) * SEQ * DMODEL + (size_t)(z % NHEAD) * DHEAD;
                   ldC = DMODEL; }

  f32x8 acc[2][2];
#pragma unroll
  for (int i = 0; i < 2; ++i)
#pragma unroll
    for (int j = 0; j < 2; ++j) acc[i][j] = (f32x8){0.f,0.f,0.f,0.f,0.f,0.f,0.f,0.f};

  const int khalf = (lane >> 4) * 8;
  const int koff  = (lane >> 4) * 8;
  const int cl    = lane & 15;

  auto load_tile = [&](int kk, int buf_) {
    *(uint4*)&As[buf_][lra][lca]     = *(const uint4*)(A  + (size_t)(m0 + lra) * K + kk + lca);
    *(uint4*)&As[buf_][lra][lca + 8] = *(const uint4*)(A  + (size_t)(m0 + lra) * K + kk + lca + 8);
    *(uint4*)&Bs[buf_][lrb][lcb]     = *(const uint4*)(Bt + (size_t)(n0 + lrb) * K + kk + lcb);
  };

  int buf = 0;
  load_tile(0, 0);
  __syncthreads();

  for (int kk = 0; kk < K; kk += 32) {
    const int nx = buf ^ 1;
    if (kk + 32 < K) load_tile(kk + 32, nx);
    bf16x16 af[2];
#pragma unroll
    for (int i = 0; i < 2; ++i) {
      const int am = wm * 32 + i * 16 + cl;
      bf16x8 lo = *(const bf16x8*)&As[buf][am][khalf];
      bf16x8 hi = *(const bf16x8*)&As[buf][am][16 + khalf];
      af[i] = SHUF16(lo, hi);
    }
    bf16x16 bfr[2];
#pragma unroll
    for (int j = 0; j < 2; ++j) {
      const int bn = wn * 32 + j * 16 + cl;
      bf16x8 lo = *(const bf16x8*)&Bs[buf][bn][koff];
      bf16x8 hi = *(const bf16x8*)&Bs[buf][bn][koff + 16];
      bfr[j] = SHUF16(lo, hi);
    }
#pragma unroll
    for (int i = 0; i < 2; ++i)
#pragma unroll
      for (int j = 0; j < 2; ++j)
        acc[i][j] = __builtin_amdgcn_wmma_f32_16x16x32_bf16(false, af[i], false, bfr[j],
                                                            (short)0, acc[i][j], false, false);
    __syncthreads();
    buf = nx;
  }

  const int rbase = (lane >> 4) << 3;
#pragma unroll
  for (int i = 0; i < 2; ++i)
#pragma unroll
    for (int j = 0; j < 2; ++j) {
      const int mrow = m0 + wm * 32 + i * 16 + rbase;
      const int ncol = n0 + wn * 32 + j * 16 + cl;
#pragma unroll
      for (int r = 0; r < 8; ++r)
        C[(size_t)(mrow + r) * ldC + ncol] = acc[i][j][r];
    }
}

// -------- masked softmax over one score row; emits bf16 probabilities --------
__global__ __launch_bounds__(256) void softmax_kernel(const float* __restrict__ scores,
                                                      const int* __restrict__ amask,
                                                      bf16_t* __restrict__ Pbf) {
  __shared__ float red[256];
  const int z = blockIdx.y, q = blockIdx.x, tid = threadIdx.x;
  const float* srow = scores + ((size_t)z * SEQ + q) * SEQ;
  const int*   mrow = amask + (z / NHEAD) * SEQ;
  float v[2];
  float mx = -3.0e38f;
#pragma unroll
  for (int j = 0; j < 2; ++j) {
    const int k = tid + j * 256;
    float s = (k <= q) ? srow[k] : -10000.0f;         // w*c - 10000*(1-c)
    s += (1.0f - (float)mrow[k]) * -10000.0f;         // additive attention mask
    v[j] = s;
    mx = fmaxf(mx, s);
  }
  red[tid] = mx; __syncthreads();
  for (int o = 128; o > 0; o >>= 1) { if (tid < o) red[tid] = fmaxf(red[tid], red[tid + o]); __syncthreads(); }
  mx = red[0]; __syncthreads();
  float sum = 0.f;
#pragma unroll
  for (int j = 0; j < 2; ++j) { v[j] = expf(v[j] - mx); sum += v[j]; }
  red[tid] = sum; __syncthreads();
  for (int o = 128; o > 0; o >>= 1) { if (tid < o) red[tid] += red[tid + o]; __syncthreads(); }
  const float inv = 1.0f / red[0];
  bf16_t* prow = Pbf + ((size_t)z * SEQ + q) * SEQ;
#pragma unroll
  for (int j = 0; j < 2; ++j) prow[tid + j * 256] = (bf16_t)(v[j] * inv);
}

// ---------------- out = LayerNorm(x + y) * g + b ----------------
__global__ __launch_bounds__(256) void add_ln_kernel(const float* __restrict__ x,
                                                     const float* __restrict__ y,
                                                     const float* __restrict__ g,
                                                     const float* __restrict__ bb,
                                                     float* __restrict__ out) {
  __shared__ float red[256];
  const int t = blockIdx.x, tid = threadIdx.x;
  float v[3];
  float s = 0.f;
#pragma unroll
  for (int j = 0; j < 3; ++j) {
    int i = tid + j * 256;
    float xv = x[(size_t)t * DMODEL + i] + y[(size_t)t * DMODEL + i];
    v[j] = xv; s += xv;
  }
  red[tid] = s; __syncthreads();
  for (int o = 128; o > 0; o >>= 1) { if (tid < o) red[tid] += red[tid + o]; __syncthreads(); }
  const float mean = red[0] / DMODEL;
  __syncthreads();
  float ss = 0.f;
#pragma unroll
  for (int j = 0; j < 3; ++j) { float d = v[j] - mean; ss += d * d; }
  red[tid] = ss; __syncthreads();
  for (int o = 128; o > 0; o >>= 1) { if (tid < o) red[tid] += red[tid + o]; __syncthreads(); }
  const float rstd = rsqrtf(red[0] / DMODEL + LN_EPS);
#pragma unroll
  for (int j = 0; j < 3; ++j) {
    int i = tid + j * 256;
    out[(size_t)t * DMODEL + i] = (v[j] - mean) * rstd * g[i] + bb[i];
  }
}

__global__ __launch_bounds__(256) void copy_kernel(const float* __restrict__ src,
                                                   float* __restrict__ dst, size_t n) {
  size_t i = (size_t)blockIdx.x * 256 + threadIdx.x;
  if (i < n) dst[i] = src[i];
}

// ---------------- host orchestration ----------------
static inline size_t alignup(size_t v) { return (v + 255) & ~(size_t)255; }

extern "C" void kernel_launch(void* const* d_in, const int* in_sizes, int n_in,
                              void* d_out, int out_size, void* d_ws, size_t ws_size,
                              hipStream_t stream) {
  (void)in_sizes; (void)n_in; (void)out_size; (void)ws_size;
  const int*   ids   = (const int*)d_in[0];
  const int*   amask = (const int*)d_in[1];
  const float* tok   = (const float*)d_in[2];
  const float* pos   = (const float*)d_in[3];
  const float* Wqkv  = (const float*)d_in[4];
  const float* bqkv  = (const float*)d_in[5];
  const float* Wo    = (const float*)d_in[6];
  const float* bo    = (const float*)d_in[7];
  const float* ln1g  = (const float*)d_in[8];
  const float* ln1b  = (const float*)d_in[9];
  const float* Wfc   = (const float*)d_in[10];
  const float* bfc   = (const float*)d_in[11];
  const float* Wp2   = (const float*)d_in[12];
  const float* bp2   = (const float*)d_in[13];
  const float* ln2g  = (const float*)d_in[14];
  const float* ln2b  = (const float*)d_in[15];

  const size_t TD       = (size_t)TOKENS * DMODEL;
  const size_t qkvBytes = alignup((size_t)TOKENS * QKVD * 4);
  const size_t ffBytes  = alignup((size_t)TOKENS * DFF * 4);
  const size_t tmpBytes = alignup(TD * 4);
  const size_t scBytes  = alignup((size_t)BH * SEQ * SEQ * 4);
  const size_t uBytes   = (qkvBytes + ffBytes + tmpBytes) > scBytes
                          ? (qkvBytes + ffBytes + tmpBytes) : scBytes;

  char* w = (char*)d_ws;
  float*  h    = (float*)w; w += alignup(TD * 4);
  float*  a    = (float*)w; w += alignup(TD * 4);
  float*  nbuf = (float*)w; w += alignup(TD * 4);
  // union region: {qkv | ff | tmp} alias {scores}; never live simultaneously.
  char* U = w; w += alignup(uBytes);
  float* qkv    = (float*)U;
  float* ff     = (float*)(U + qkvBytes);
  float* tmp    = (float*)(U + qkvBytes + ffBytes);
  float* scores = (float*)U;
  bf16_t* Pbf   = (bf16_t*)w; w += alignup((size_t)BH * SEQ * SEQ * 2);
  bf16_t* Qbf   = (bf16_t*)w; w += alignup((size_t)BH * SEQ * DHEAD * 2);
  bf16_t* Kbf   = (bf16_t*)w; w += alignup((size_t)BH * SEQ * DHEAD * 2);
  bf16_t* Vtbf  = (bf16_t*)w; w += alignup((size_t)BH * DHEAD * SEQ * 2);
  bf16_t* actbf = (bf16_t*)w; w += alignup((size_t)TOKENS * DFF * 2);
  bf16_t* wqkvbf = (bf16_t*)w; w += alignup((size_t)NLAYER * DMODEL * QKVD * 2);
  bf16_t* wobf   = (bf16_t*)w; w += alignup((size_t)NLAYER * DMODEL * DMODEL * 2);
  bf16_t* wfcbf  = (bf16_t*)w; w += alignup((size_t)NLAYER * DMODEL * DFF * 2);
  bf16_t* wp2bf  = (bf16_t*)w; w += alignup((size_t)NLAYER * DFF * DMODEL * 2);

  auto blocks = [](size_t n) { return (unsigned)((n + 255) / 256); };

  // weight precision conversion + transpose to [N,K] (once per call)
  cvt_w_t_kernel<<<blocks((size_t)NLAYER * DMODEL * QKVD), 256, 0, stream>>>(Wqkv, wqkvbf, DMODEL, QKVD, NLAYER);
  cvt_w_t_kernel<<<blocks((size_t)NLAYER * DMODEL * DMODEL), 256, 0, stream>>>(Wo, wobf, DMODEL, DMODEL, NLAYER);
  cvt_w_t_kernel<<<blocks((size_t)NLAYER * DMODEL * DFF), 256, 0, stream>>>(Wfc, wfcbf, DMODEL, DFF, NLAYER);
  cvt_w_t_kernel<<<blocks((size_t)NLAYER * DFF * DMODEL), 256, 0, stream>>>(Wp2, wp2bf, DFF, DMODEL, NLAYER);

  embed_kernel<<<blocks(TD), 256, 0, stream>>>(ids, tok, pos, h);

  for (int l = 0; l < NLAYER; ++l) {
    // ---- QKV projection ----
    cvt_bf16_kernel<<<blocks(TD), 256, 0, stream>>>(h, actbf, TD);
    gemm_bf16_wmma<0><<<dim3(QKVD / 128, TOKENS / 128), 256, 0, stream>>>(
        actbf, wqkvbf + (size_t)l * DMODEL * QKVD, bqkv + (size_t)l * QKVD, qkv,
        TOKENS, QKVD, DMODEL);
    // ---- attention (all WMMA) ----
    qkv_split_kernel<<<blocks(TD), 256, 0, stream>>>(qkv, Qbf, Kbf, Vtbf);
    attn_gemm_wmma<0><<<dim3(SEQ / 64, SEQ / 128, BH), 256, 0, stream>>>(
        Qbf, Kbf, scores, SEQ, SEQ, DHEAD);
    softmax_kernel<<<dim3(SEQ, BH), 256, 0, stream>>>(scores, amask, Pbf);
    attn_gemm_wmma<1><<<dim3(DHEAD / 64, SEQ / 128, BH), 256, 0, stream>>>(
        Pbf, Vtbf, a, SEQ, DHEAD, SEQ);
    // ---- output projection ----
    cvt_bf16_kernel<<<blocks(TD), 256, 0, stream>>>(a, actbf, TD);
    gemm_bf16_wmma<0><<<dim3(DMODEL / 128, TOKENS / 128), 256, 0, stream>>>(
        actbf, wobf + (size_t)l * DMODEL * DMODEL, bo + (size_t)l * DMODEL, tmp,
        TOKENS, DMODEL, DMODEL);
    // ---- n = LN(h + attn_proj) ----
    add_ln_kernel<<<TOKENS, 256, 0, stream>>>(h, tmp, ln1g + (size_t)l * DMODEL,
                                              ln1b + (size_t)l * DMODEL, nbuf);
    // ---- FFN up + GELU ----
    cvt_bf16_kernel<<<blocks(TD), 256, 0, stream>>>(nbuf, actbf, TD);
    gemm_bf16_wmma<1><<<dim3(DFF / 128, TOKENS / 128), 256, 0, stream>>>(
        actbf, wfcbf + (size_t)l * DMODEL * DFF, bfc + (size_t)l * DFF, ff,
        TOKENS, DFF, DMODEL);
    // ---- FFN down ----
    cvt_bf16_kernel<<<blocks((size_t)TOKENS * DFF), 256, 0, stream>>>(ff, actbf, (size_t)TOKENS * DFF);
    gemm_bf16_wmma<0><<<dim3(DMODEL / 128, TOKENS / 128), 256, 0, stream>>>(
        actbf, wp2bf + (size_t)l * DFF * DMODEL, bp2 + (size_t)l * DMODEL, tmp,
        TOKENS, DMODEL, DFF);
    // ---- h = LN(n + ffn) ----
    add_ln_kernel<<<TOKENS, 256, 0, stream>>>(nbuf, tmp, ln2g + (size_t)l * DMODEL,
                                              ln2b + (size_t)l * DMODEL, h);
  }

  copy_kernel<<<blocks(TD), 256, 0, stream>>>(h, (float*)d_out, TD);
}
